// GRU2_55164559949999
// MI455X (gfx1250) — matrix operand order repaired and checked by reading, not compiled, not run on hardware
//
#include <hip/hip_runtime.h>
#include <math.h>

// ---------------- Problem constants ----------------
#define BB   32
#define TT   512
#define HH   512
#define VV   10000
#define MROWS (TT*BB)        // 16384

typedef __attribute__((ext_vector_type(16))) __bf16 v16bf;
typedef __attribute__((ext_vector_type(8)))  __bf16 v8bf;
typedef __attribute__((ext_vector_type(8)))  float  v8f;
typedef __attribute__((ext_vector_type(4)))  unsigned int u32x4;
typedef __attribute__((ext_vector_type(8)))  unsigned int u32x8;

union BF16x16 { v16bf v; struct { v8bf lo; v8bf hi; } p; };

__device__ __forceinline__ v8f wmma_bf16(const BF16x16& a, const BF16x16& b, v8f c) {
    return __builtin_amdgcn_wmma_f32_16x16x32_bf16(false, a.v, false, b.v,
                                                   (short)0, c, false, false);
}

// ---- Tensor Data Mover: 2D tile (tile_d0 x tile_d1, 2-byte elems) -> LDS ----
// D# per CDNA5 ISA. Rows beyond tensor_d1 are zero-filled.
__device__ __forceinline__ void tdm_load_tile_2d(
    unsigned lds_off, const void* gaddr,
    unsigned tensor_d0, unsigned tensor_d1,
    unsigned tile_d0, unsigned tile_d1,
    unsigned long long stride0_elems)
{
    unsigned long long ga = (unsigned long long)gaddr;
    u32x4 g0;
    g0[0] = 1u;                                        // count=1, no gather
    g0[1] = lds_off;                                   // lds_addr
    g0[2] = (unsigned)(ga & 0xFFFFFFFFu);              // global_addr[31:0]
    g0[3] = (unsigned)((ga >> 32) & 0x01FFFFFFu)       // global_addr[56:32]
          | (2u << 30);                                // type=2 ("image")
    u32x8 g1;
    g1[0] = 0x00010000u;                               // wg_mask=0, data_size=1 (2B)
    g1[1] = (tensor_d0 & 0xFFFFu) << 16;               // tensor_dim0[15:0]
    g1[2] = (tensor_d0 >> 16) | ((tensor_d1 & 0xFFFFu) << 16);
    g1[3] = ((tensor_d1 >> 16) & 0xFFFFu) | (tile_d0 << 16);
    g1[4] = tile_d1 & 0xFFFFu;                         // tile_dim1, tile_dim2=0
    g1[5] = (unsigned)(stride0_elems & 0xFFFFFFFFu);   // tensor_dim0_stride
    g1[6] = (unsigned)((stride0_elems >> 32) & 0xFFFFu);
    g1[7] = 0u;
    asm volatile("tensor_load_to_lds %0, %1" :: "s"(g0), "s"(g1) : "memory");
}

// ---------------- embed gather + cast ----------------
__global__ __launch_bounds__(256) void embed_cast_kernel(
    const int* __restrict__ x, const float* __restrict__ Wemb,
    __bf16* __restrict__ Xb)
{
    int row = blockIdx.x;                 // row = t*B + b
    int t = row >> 5, b = row & 31;
    int tok = x[b * TT + t];
    const float* src = Wemb + (size_t)tok * HH;
    __bf16* dst = Xb + (size_t)row * HH;
    for (int h = threadIdx.x; h < HH; h += blockDim.x)
        dst[h] = (__bf16)src[h];
}

// ---------------- weight split + cast ----------------
__global__ __launch_bounds__(256) void cast_weights_kernel(
    const float* __restrict__ Wz, const float* __restrict__ Wr,
    const float* __restrict__ Wh, const float* __restrict__ Wd,
    __bf16* __restrict__ Wzh, __bf16* __restrict__ Wrh, __bf16* __restrict__ Whh,
    __bf16* __restrict__ Wzx, __bf16* __restrict__ Wrx, __bf16* __restrict__ Whx,
    __bf16* __restrict__ Wdb)
{
    int stride = gridDim.x * blockDim.x;
    for (int i = blockIdx.x * blockDim.x + threadIdx.x; i < HH * HH; i += stride) {
        int j = i >> 9, k = i & 511;
        Wzh[i] = (__bf16)Wz[j * (2*HH) + k];
        Wzx[i] = (__bf16)Wz[j * (2*HH) + HH + k];
        Wrh[i] = (__bf16)Wr[j * (2*HH) + k];
        Wrx[i] = (__bf16)Wr[j * (2*HH) + HH + k];
        Whh[i] = (__bf16)Wh[j * (2*HH) + k];
        Whx[i] = (__bf16)Wh[j * (2*HH) + HH + k];
    }
    for (int i = blockIdx.x * blockDim.x + threadIdx.x; i < VV * HH; i += stride)
        Wdb[i] = (__bf16)Wd[i];
}

// ------- TN GEMM: C[M,N](f32) = A[M,K]bf16 @ B[N,K]^T bf16 (+bias, +row permute) -------
// 128x128 WG tile, TDM-staged double-buffered LDS panels, 32x64 per wave (2x4 WMMA).
#define WG_M 128
#define WG_N 128
#define KC   32

__global__ __launch_bounds__(256) void gemm_tn_bf16_kernel(
    int M, int N, int K,
    const __bf16* __restrict__ A, const __bf16* __restrict__ Bm,
    float* __restrict__ C, int ldc,
    const float* __restrict__ bias, int permute)
{
    __shared__ __align__(16) __bf16 As[2][WG_M][KC];   // 2 x 8 KB
    __shared__ __align__(16) __bf16 Bs[2][WG_N][KC];   // 2 x 8 KB

    const int tid  = threadIdx.x;
    const int wave = tid >> 5;
    const int lane = tid & 31;
    const int m0 = blockIdx.y * WG_M;
    const int n0 = blockIdx.x * WG_N;
    const int wm = (wave >> 1) * 32;     // 0,32,64,96
    const int wn = (wave & 1) * 64;      // 0,64
    const int c16 = lane & 15;
    const int akb = (lane < 16) ? 0 : 8;
    const int bkb = (lane < 16) ? 0 : 8;

    v8f acc[2][4];
    for (int i = 0; i < 2; i++)
        for (int j = 0; j < 4; j++)
            for (int e = 0; e < 8; e++) acc[i][j][e] = 0.0f;

    const int iters = K / KC;

    // prologue: DMA first panels into buffer 0
    if (wave == 0) {
        tdm_load_tile_2d((unsigned)(uintptr_t)&As[0][0][0], A + (size_t)m0 * K,
                         (unsigned)K, (unsigned)(M - m0), KC, WG_M, (unsigned long long)K);
        tdm_load_tile_2d((unsigned)(uintptr_t)&Bs[0][0][0], Bm + (size_t)n0 * K,
                         (unsigned)K, (unsigned)(N - n0), KC, WG_N, (unsigned long long)K);
        __builtin_amdgcn_s_wait_tensorcnt(0);
    }
    __syncthreads();

    for (int it = 0; it < iters; ++it) {
        const int cur = it & 1;
        // kick off DMA for next K-slice into the other buffer (overlaps WMMAs)
        if (wave == 0 && it + 1 < iters) {
            const int k0n = (it + 1) * KC;
            tdm_load_tile_2d((unsigned)(uintptr_t)&As[cur ^ 1][0][0],
                             A + (size_t)m0 * K + k0n,
                             (unsigned)(K - k0n), (unsigned)(M - m0), KC, WG_M,
                             (unsigned long long)K);
            tdm_load_tile_2d((unsigned)(uintptr_t)&Bs[cur ^ 1][0][0],
                             Bm + (size_t)n0 * K + k0n,
                             (unsigned)(K - k0n), (unsigned)(N - n0), KC, WG_N,
                             (unsigned long long)K);
        }

        BF16x16 af[2], bfr[4];
        for (int i = 0; i < 2; i++) {
            const __bf16* ap = &As[cur][wm + i * 16 + c16][0];
            af[i].p.lo = *(const v8bf*)(ap + akb);
            af[i].p.hi = *(const v8bf*)(ap + akb + 16);
        }
        for (int j = 0; j < 4; j++) {
            const __bf16* bp = &Bs[cur][wn + j * 16 + c16][0];
            bfr[j].p.lo = *(const v8bf*)(bp + bkb);
            bfr[j].p.hi = *(const v8bf*)(bp + bkb + 16);
        }
        for (int i = 0; i < 2; i++)
            for (int j = 0; j < 4; j++)
                acc[i][j] = wmma_bf16(af[i], bfr[j], acc[i][j]);

        if (wave == 0 && it + 1 < iters) __builtin_amdgcn_s_wait_tensorcnt(0);
        __syncthreads();
    }

    // epilogue: element (vgpr r, lane) -> row r+8*(lane>=16), col lane&15
    const int hi4 = lane >> 4;
    for (int i = 0; i < 2; i++)
        for (int j = 0; j < 4; j++) {
            int gcol = n0 + wn + j * 16 + c16;
            if (gcol >= N) continue;
            float bv = bias ? bias[gcol] : 0.0f;
            for (int r = 0; r < 8; r++) {
                int grow = m0 + wm + i * 16 + r + 8 * hi4;   // grow = t*B + b
                int orow = grow;
                if (permute) { int b = grow & 31, t = grow >> 5; orow = b * TT + t; }
                C[(size_t)orow * ldc + gcol] = acc[i][j][r] + bv;
            }
        }
}

// ---------------- persistent GRU scan with global-atomic grid barrier ----------------
#define SCAN_WGS 16

__device__ __forceinline__ void grid_barrier(int* bar, int& count) {
    __threadfence();
    __syncthreads();
    count++;
    if (threadIdx.x == 0) {
        atomicAdd(bar, 1);
        int target = count * SCAN_WGS;
        while (__hip_atomic_load(bar, __ATOMIC_RELAXED, __HIP_MEMORY_SCOPE_AGENT) < target)
            __builtin_amdgcn_s_sleep(2);
    }
    __syncthreads();
    __threadfence();
}

__global__ __launch_bounds__(256) void gru_scan_kernel(
    const __bf16* __restrict__ Wzh, const __bf16* __restrict__ Wrh,
    const __bf16* __restrict__ Whh,
    const float* __restrict__ Az, const float* __restrict__ Ar,
    const float* __restrict__ Ah,
    float* hf0, float* hf1, __bf16* hb0, __bf16* hb1,
    float* __restrict__ zbuf, __bf16* __restrict__ rhbuf,
    __bf16* __restrict__ hseq, float* __restrict__ hlast, int* bar)
{
    const int tid = threadIdx.x;
    const int gthread = blockIdx.x * 256 + tid;
    const int gw = gthread >> 5;          // global wave id 0..127
    const int lane = tid & 31;
    const int c16 = lane & 15;
    const int akb = (lane < 16) ? 0 : 8;
    const int bkb = (lane < 16) ? 0 : 8;
    int count = 0;

    // h_0 = 0
    for (int i = gthread; i < BB * HH; i += SCAN_WGS * 256) {
        hf0[i] = 0.0f; hb0[i] = (__bf16)0.0f;
    }
    grid_barrier(bar, count);

    for (int t = 0; t < TT; t++) {
        const float*  hfr = (t & 1) ? hf1 : hf0;
        float*        hfw = (t & 1) ? hf0 : hf1;
        const __bf16* hbr = (t & 1) ? hb1 : hb0;
        __bf16*       hbw = (t & 1) ? hb0 : hb1;

        // ---- phase A: z (waves 0..63) and r (waves 64..127), one 16x16 tile each
        {
            int g = gw >> 6;                       // 0 = z, 1 = r
            int ti = gw & 63;
            int m0 = (ti >> 5) * 16, n0 = (ti & 31) * 16;
            const __bf16* W  = g ? Wrh : Wzh;
            const float*  Ag = g ? Ar  : Az;

            v8f acc; for (int e = 0; e < 8; e++) acc[e] = 0.0f;
            const __bf16* arow = hbr + (size_t)(m0 + c16) * HH;
            const __bf16* brow = W   + (size_t)(n0 + c16) * HH;
            for (int kc = 0; kc < HH; kc += 32) {
                BF16x16 a, b;
                a.p.lo = *(const v8bf*)(arow + kc + akb);
                a.p.hi = *(const v8bf*)(arow + kc + akb + 16);
                b.p.lo = *(const v8bf*)(brow + kc + bkb);
                b.p.hi = *(const v8bf*)(brow + kc + bkb + 16);
                acc = wmma_bf16(a, b, acc);
            }
            int hi4 = lane >> 4;
            for (int r = 0; r < 8; r++) {
                int row = m0 + r + 8 * hi4;        // batch index b
                int col = n0 + c16;                // hidden unit j
                int idx = row * HH + col;
                float pre = acc[r] + Ag[(size_t)(t * BB + row) * HH + col];
                float s = 1.0f / (1.0f + expf(-pre));
                if (g == 0) zbuf[idx] = s;
                else        rhbuf[idx] = (__bf16)(s * hfr[idx]);
            }
        }
        grid_barrier(bar, count);

        // ---- phase B: h_tilde + h_new, waves 0..63
        if (gw < 64) {
            int ti = gw;
            int m0 = (ti >> 5) * 16, n0 = (ti & 31) * 16;
            v8f acc; for (int e = 0; e < 8; e++) acc[e] = 0.0f;
            const __bf16* arow = rhbuf + (size_t)(m0 + c16) * HH;
            const __bf16* brow = Whh   + (size_t)(n0 + c16) * HH;
            for (int kc = 0; kc < HH; kc += 32) {
                BF16x16 a, b;
                a.p.lo = *(const v8bf*)(arow + kc + akb);
                a.p.hi = *(const v8bf*)(arow + kc + akb + 16);
                b.p.lo = *(const v8bf*)(brow + kc + bkb);
                b.p.hi = *(const v8bf*)(brow + kc + bkb + 16);
                acc = wmma_bf16(a, b, acc);
            }
            int hi4 = lane >> 4;
            for (int r = 0; r < 8; r++) {
                int row = m0 + r + 8 * hi4;
                int col = n0 + c16;
                int idx = row * HH + col;
                float pre = acc[r] + Ah[(size_t)(t * BB + row) * HH + col];
                float ht = tanhf(pre);
                float z  = zbuf[idx];
                float hv = hfr[idx];
                float hn = (1.0f - z) * hv + z * ht;
                hfw[idx] = hn;
                __bf16 hb = (__bf16)hn;
                hbw[idx] = hb;
                hseq[(size_t)(t * BB + row) * HH + col] = hb;
            }
        }
        grid_barrier(bar, count);
    }

    // final h lives in buffer 0 (t=511 writes hf0)
    for (int i = gthread; i < BB * HH; i += SCAN_WGS * 256)
        hlast[i] = hf0[i];
}

// ---------------- host launch ----------------
extern "C" void kernel_launch(void* const* d_in, const int* in_sizes, int n_in,
                              void* d_out, int out_size, void* d_ws, size_t ws_size,
                              hipStream_t stream) {
    const int*   x    = (const int*)  d_in[0];
    const float* Wemb = (const float*)d_in[1];
    const float* Wz   = (const float*)d_in[2];
    const float* Wr   = (const float*)d_in[3];
    const float* Wh   = (const float*)d_in[4];
    const float* Wd   = (const float*)d_in[5];
    const float* bd   = (const float*)d_in[6];

    float* out   = (float*)d_out;                       // [B,T,V]
    float* hlast = out + (size_t)BB * TT * VV;          // [B,H]

    char* p = (char*)d_ws;
    auto alloc = [&](size_t bytes) -> char* {
        char* r = p; p += (bytes + 255) & ~(size_t)255; return r;
    };
    __bf16* Xb   = (__bf16*)alloc((size_t)MROWS * HH * 2);
    __bf16* Wzh  = (__bf16*)alloc((size_t)HH * HH * 2);
    __bf16* Wrh  = (__bf16*)alloc((size_t)HH * HH * 2);
    __bf16* Whh  = (__bf16*)alloc((size_t)HH * HH * 2);
    __bf16* Wzx  = (__bf16*)alloc((size_t)HH * HH * 2);
    __bf16* Wrx  = (__bf16*)alloc((size_t)HH * HH * 2);
    __bf16* Whx  = (__bf16*)alloc((size_t)HH * HH * 2);
    __bf16* Wdb  = (__bf16*)alloc((size_t)VV * HH * 2);
    float*  Az   = (float*) alloc((size_t)MROWS * HH * 4);
    float*  Ar   = (float*) alloc((size_t)MROWS * HH * 4);
    float*  Ah   = (float*) alloc((size_t)MROWS * HH * 4);
    __bf16* hseq = (__bf16*)alloc((size_t)MROWS * HH * 2);
    float*  hf0  = (float*) alloc((size_t)BB * HH * 4);
    float*  hf1  = (float*) alloc((size_t)BB * HH * 4);
    __bf16* hb0  = (__bf16*)alloc((size_t)BB * HH * 2);
    __bf16* hb1  = (__bf16*)alloc((size_t)BB * HH * 2);
    float*  zbuf = (float*) alloc((size_t)BB * HH * 4);
    __bf16* rhbuf= (__bf16*)alloc((size_t)BB * HH * 2);
    int*    bar  = (int*)   alloc(256);

    hipMemsetAsync(bar, 0, 256, stream);

    // 1) embedding gather + bf16 cast
    embed_cast_kernel<<<MROWS, 256, 0, stream>>>(x, Wemb, Xb);

    // 2) weight split/cast
    cast_weights_kernel<<<2048, 256, 0, stream>>>(Wz, Wr, Wh, Wd,
                                                  Wzh, Wrh, Whh, Wzx, Wrx, Whx, Wdb);

    // 3) input projections for all timesteps (parallel GEMMs)
    dim3 gP(HH / WG_N, MROWS / WG_M);
    gemm_tn_bf16_kernel<<<gP, 256, 0, stream>>>(MROWS, HH, HH, Xb, Wzx, Az, HH, nullptr, 0);
    gemm_tn_bf16_kernel<<<gP, 256, 0, stream>>>(MROWS, HH, HH, Xb, Wrx, Ar, HH, nullptr, 0);
    gemm_tn_bf16_kernel<<<gP, 256, 0, stream>>>(MROWS, HH, HH, Xb, Whx, Ah, HH, nullptr, 0);

    // 4) sequential recurrence (persistent, grid-barrier synced)
    gru_scan_kernel<<<SCAN_WGS, 256, 0, stream>>>(Wzh, Wrh, Whh, Az, Ar, Ah,
                                                  hf0, hf1, hb0, hb1,
                                                  zbuf, rhbuf, hseq, hlast, bar);

    // 5) vocabulary decode for every step: out[b,t,:] = hseq[t*B+b,:] @ Wd^T + bd
    dim3 gD((VV + WG_N - 1) / WG_N, MROWS / WG_M);
    gemm_tn_bf16_kernel<<<gD, 256, 0, stream>>>(MROWS, VV, HH, hseq, Wdb, out, VV, bd, 1);

    (void)in_sizes; (void)n_in; (void)out_size; (void)ws_size;
}
